// ResidualSelfAttention_40664750359348
// MI455X (gfx1250) — matrix operand order repaired and checked by reading, not compiled, not run on hardware
//
#include <hip/hip_runtime.h>
#include <hip/hip_bf16.h>
#include <math.h>

// ---------------------------------------------------------------------------
// Transformer block (residual self-attention + MLP) for MI455X / gfx1250.
// All matmuls use v_wmma_f32_16x16x32_bf16 (fp32 accumulate); tile staging
// uses CDNA5 async global->LDS copies (ASYNCcnt) instead of VGPR round-trips.
// Workload is compute bound: ~137 GFLOP vs ~70MB traffic @ 23.3 TB/s.
// ---------------------------------------------------------------------------

typedef __attribute__((ext_vector_type(16))) __bf16 v16bf;
typedef __attribute__((ext_vector_type(8)))  float  v8f;

#define D_MODEL  1024
#define N_HEADS  16
#define HEAD_DIM 64
#define D_FF     4096
#define SEQ      2048
#define BATCH    2
#define ROWS     (BATCH * SEQ)   // 4096 token rows

// ---- helpers ---------------------------------------------------------------

__device__ __forceinline__ unsigned short f2bf(float f) {
  unsigned int u = __float_as_uint(f);
  unsigned int r = (u + 0x7FFFu + ((u >> 16) & 1u)) >> 16;  // RNE
  return (unsigned short)r;
}
__device__ __forceinline__ __bf16 us2bf(unsigned short u) {
  union { unsigned short s; __bf16 b; } c; c.s = u; return c.b;
}
__device__ __forceinline__ float gelu_tanh(float x) {
  float x3 = x * x * x;
  return 0.5f * x * (1.0f + tanhf(0.7978845608f * (x + 0.044715f * x3)));
}

// CDNA5 async global->LDS copies (GLOBAL_LOAD_ASYNC_TO_LDS_B128, ASYNCcnt).
// INST_OFFSET is added to BOTH the global and the LDS address per the ISA,
// so one (lds,addr) register pair covers a contiguous multi-b128 run.
__device__ __forceinline__ void async_copy_b128x2(const void* g, void* l) {
  unsigned long long ga = (unsigned long long)g;
  unsigned la = (unsigned)(unsigned long long)l;   // low 32 bits = LDS offset
  asm volatile("global_load_async_to_lds_b128 %0, %1, off\n\t"
               "global_load_async_to_lds_b128 %0, %1, off offset:16"
               :: "v"(la), "v"(ga) : "memory");
}
__device__ __forceinline__ void async_copy_b128x4(const void* g, void* l) {
  unsigned long long ga = (unsigned long long)g;
  unsigned la = (unsigned)(unsigned long long)l;
  asm volatile("global_load_async_to_lds_b128 %0, %1, off\n\t"
               "global_load_async_to_lds_b128 %0, %1, off offset:16\n\t"
               "global_load_async_to_lds_b128 %0, %1, off offset:32\n\t"
               "global_load_async_to_lds_b128 %0, %1, off offset:48"
               :: "v"(la), "v"(ga) : "memory");
}
__device__ __forceinline__ void async_wait0() {
  asm volatile("s_wait_asynccnt 0x0" ::: "memory");
}

// A-fragment (16x32 bf16, MxK) in contiguous k order (same as the B frags):
// lanes 0-15 hold M=lane; elems 0..15 -> K = half*16 + e
__device__ __forceinline__ v16bf frag_a(const unsigned short* base, int ld,
                                        int row0, int kc, int lane) {
  int m = lane & 15, half = lane >> 4;
  const unsigned short* p = base + (row0 + m) * ld + kc + half * 16;
  v16bf a;
#pragma unroll
  for (int q = 0; q < 8; q++) a[q] = us2bf(p[q]);
#pragma unroll
  for (int q = 0; q < 8; q++) a[8 + q] = us2bf(p[8 + q]);
  return a;
}

// B-fragment (32x16 bf16, KxN) from a K-major (transposed) LDS tile:
// lane n holds column N=n; lanes 0-15: K=0..15, lanes 16-31: K=16..31.
__device__ __forceinline__ v16bf frag_bT(const unsigned short* baseT, int ld,
                                         int col0, int kc, int lane) {
  int n = lane & 15, half = lane >> 4;
  const unsigned short* p = baseT + (col0 + n) * ld + kc + half * 16;
  v16bf b;
#pragma unroll
  for (int q = 0; q < 16; q++) b[q] = us2bf(p[q]);
  return b;
}

// Same B-fragment but from a row-major (K rows) LDS tile.
__device__ __forceinline__ v16bf frag_bRM(const unsigned short* base, int ld,
                                          int kc, int col0, int lane) {
  int n = lane & 15, half = lane >> 4;
  const unsigned short* p = base + (kc + half * 16) * ld + col0 + n;
  v16bf b;
#pragma unroll
  for (int q = 0; q < 16; q++) b[q] = us2bf(p[q * ld]);
  return b;
}

// ---- elementwise cast fp32 -> bf16 ----------------------------------------

__global__ __launch_bounds__(256)
void cast_f32_bf16_kernel(const float* __restrict__ X,
                          unsigned short* __restrict__ Y, int n) {
  int i = (blockIdx.x * 256 + threadIdx.x) * 4;
  if (i + 3 < n) {
    float4 v = *(const float4*)(X + i);
    Y[i + 0] = f2bf(v.x);
    Y[i + 1] = f2bf(v.y);
    Y[i + 2] = f2bf(v.z);
    Y[i + 3] = f2bf(v.w);
  } else {
    for (int j = i; j < n; j++) Y[j] = f2bf(X[j]);
  }
}

// ---- row LayerNorm (+ optional plain cast) --------------------------------

__global__ __launch_bounds__(256)
void layernorm_cast_kernel(const float* __restrict__ X,
                           const float* __restrict__ scale,
                           const float* __restrict__ bias,
                           unsigned short* __restrict__ Yn,
                           unsigned short* __restrict__ Yc) {
  __shared__ float red[256];
  const int row = blockIdx.x;
  const float* xr = X + (size_t)row * D_MODEL;

  float s = 0.f;
  for (int i = threadIdx.x; i < D_MODEL; i += 256) s += xr[i];
  red[threadIdx.x] = s; __syncthreads();
  for (int o = 128; o > 0; o >>= 1) {
    if (threadIdx.x < o) red[threadIdx.x] += red[threadIdx.x + o];
    __syncthreads();
  }
  float mu = red[0] * (1.0f / D_MODEL);
  __syncthreads();

  float vs = 0.f;
  for (int i = threadIdx.x; i < D_MODEL; i += 256) {
    float d = xr[i] - mu; vs += d * d;
  }
  red[threadIdx.x] = vs; __syncthreads();
  for (int o = 128; o > 0; o >>= 1) {
    if (threadIdx.x < o) red[threadIdx.x] += red[threadIdx.x + o];
    __syncthreads();
  }
  float rstd = rsqrtf(red[0] * (1.0f / D_MODEL) + 1e-6f);

  for (int i = threadIdx.x; i < D_MODEL; i += 256) {
    float xv = xr[i];
    Yn[(size_t)row * D_MODEL + i] = f2bf((xv - mu) * rstd * scale[i] + bias[i]);
    if (Yc) Yc[(size_t)row * D_MODEL + i] = f2bf(xv);
  }
}

// ---- generic bf16 WMMA GEMM: C = act(A@B + bias) + resid ------------------
// 128x128 block tile, 256 threads (8 waves). Wave grid 4(M)x2(N):
// each wave owns a 32x64 region = 2x4 16x16 WMMA tiles.
// A tile staged with async global->LDS; B tile staged transposed (manual,
// since the async path cannot transpose).

__global__ __launch_bounds__(256)
void gemm_bf16_kernel(const unsigned short* __restrict__ A,
                      const unsigned short* __restrict__ Bm,
                      int M, int N, int K,
                      const float* __restrict__ bias,
                      const float* __restrict__ resid,
                      float* __restrict__ Cf,
                      unsigned short* __restrict__ Cb,
                      int act) {
  __shared__ unsigned short ldsA[128][40];   // 128 x 32 (+pad), M-major
  __shared__ unsigned short ldsB[128][40];   // 128 x 32 (+pad), N-major (B^T)

  const int tid  = threadIdx.x;
  const int lane = tid & 31;
  const int wid  = tid >> 5;
  const int waveM = wid & 3;    // 32 rows each
  const int waveN = wid >> 2;   // 64 cols each
  const int half = lane >> 4;
  const int ln16 = lane & 15;

  const int bm = blockIdx.y * 128;
  const int bn = blockIdx.x * 128;

  const v8f vzero = {};
  v8f acc[2][4];
#pragma unroll
  for (int i = 0; i < 2; i++)
#pragma unroll
    for (int j = 0; j < 4; j++) acc[i][j] = vzero;

  const int ra = tid >> 1, ca = (tid & 1) * 16;   // A tile loader coords
  const int rb = tid >> 3, cb = (tid & 7) * 16;   // B tile loader coords

  for (int k0 = 0; k0 < K; k0 += 32) {
    __syncthreads();
    // stage A tile (128x32) row-major: async copy, no VGPR round-trip
    async_copy_b128x2(A + (size_t)(bm + ra) * K + k0 + ca, &ldsA[ra][ca]);
    // stage B tile (32x128) transposed into ldsB[n][k]
    {
      const uint4* g = (const uint4*)(Bm + (size_t)(k0 + rb) * N + bn + cb);
      union { uint4 v; unsigned short s[8]; } u0, u1;
      u0.v = g[0]; u1.v = g[1];
#pragma unroll
      for (int j = 0; j < 8; j++) ldsB[cb + j][rb]     = u0.s[j];
#pragma unroll
      for (int j = 0; j < 8; j++) ldsB[cb + 8 + j][rb] = u1.s[j];
    }
    if (k0 + 32 < K) {  // hint next tiles into cache (global_prefetch_b8)
      __builtin_prefetch(A + (size_t)(bm + ra) * K + k0 + 32 + ca, 0, 1);
      __builtin_prefetch(Bm + (size_t)(k0 + 32 + rb) * N + bn + cb, 0, 1);
    }
    async_wait0();       // async LDS writes visible before the barrier
    __syncthreads();

    v16bf af[2], bfv[4];
#pragma unroll
    for (int i = 0; i < 2; i++)
      af[i] = frag_a(&ldsA[0][0], 40, waveM * 32 + i * 16, 0, lane);
#pragma unroll
    for (int j = 0; j < 4; j++)
      bfv[j] = frag_bT(&ldsB[0][0], 40, waveN * 64 + j * 16, 0, lane);

#pragma unroll
    for (int i = 0; i < 2; i++)
#pragma unroll
      for (int j = 0; j < 4; j++)
        acc[i][j] = __builtin_amdgcn_wmma_f32_16x16x32_bf16(
            false, af[i], false, bfv[j], (short)0, acc[i][j], false, false);
  }

  // epilogue: bias -> activation -> residual -> store (f32 and/or bf16)
#pragma unroll
  for (int i = 0; i < 2; i++) {
#pragma unroll
    for (int j = 0; j < 4; j++) {
#pragma unroll
      for (int r = 0; r < 8; r++) {
        int row = bm + waveM * 32 + i * 16 + half * 8 + r;
        int col = bn + waveN * 64 + j * 16 + ln16;
        float v = acc[i][j][r];
        if (bias)  v += bias[col];
        if (act == 1) v = gelu_tanh(v);
        if (resid) v += resid[(size_t)row * N + col];
        if (Cf) Cf[(size_t)row * N + col] = v;
        if (Cb) Cb[(size_t)row * N + col] = f2bf(v);
      }
    }
  }
}

// ---- flash attention: softmax(Q K^T * s) V per (b, h, 128-query block) ----
// 256 threads = 8 waves; wave w owns query rows [w*16, w*16+16).
// Stream 64-key blocks: QK^T (WMMA) -> online softmax -> PV (WMMA).
// Q/K/V tiles staged with async global->LDS copies.

__global__ __launch_bounds__(256)
void flash_attn_kernel(const unsigned short* __restrict__ Q,
                       const unsigned short* __restrict__ Kt,
                       const unsigned short* __restrict__ V,
                       unsigned short* __restrict__ O,
                       const float* __restrict__ sscale) {
  __shared__ unsigned short Qs[128][72];
  __shared__ unsigned short Ks[64][72];
  __shared__ unsigned short Vs[64][72];
  __shared__ float          Ss[128][68];
  __shared__ unsigned short Ps[128][72];
  __shared__ float rowmax[128], rowsum[128], rowalpha[128];

  const int tid  = threadIdx.x;
  const int lane = tid & 31;
  const int wid  = tid >> 5;
  const int half = lane >> 4;
  const int ln16 = lane & 15;

  const int blk  = blockIdx.x;          // b*256 + h*16 + qblk
  const int qblk = blk & 15;
  const int h    = (blk >> 4) & 15;
  const int b    = blk >> 8;
  const int qbase = qblk * 128;
  const size_t base = ((size_t)b * SEQ) * D_MODEL + (size_t)h * HEAD_DIM;

  // logits/sqrt(hd) then * sscale*log(L)  ->  combined scale
  const float scale = sscale[0] * logf((float)SEQ) * 0.125f;

  // stage Q tile (128x64): async copy (waited inside first loop iteration)
  {
    int r = tid >> 1, c = (tid & 1) * 32;
    async_copy_b128x4(Q + base + (size_t)(qbase + r) * D_MODEL + c, &Qs[r][c]);
  }
  if (tid < 128) { rowmax[tid] = -3.0e38f; rowsum[tid] = 0.f; }

  const v8f vzero = {};
  v8f oacc[4];
#pragma unroll
  for (int j = 0; j < 4; j++) oacc[j] = vzero;

  const int rk = tid >> 2, ck = (tid & 3) * 16;   // K/V tile loader coords

  for (int kb = 0; kb < SEQ; kb += 64) {
    __syncthreads();   // prior PV done before K/V overwrite
    async_copy_b128x2(Kt + base + (size_t)(kb + rk) * D_MODEL + ck, &Ks[rk][ck]);
    async_copy_b128x2(V  + base + (size_t)(kb + rk) * D_MODEL + ck, &Vs[rk][ck]);
    async_wait0();
    __syncthreads();

    // S = (Q K^T) * scale  (128x64; wave rows x 4 key tiles; hd=64 -> 2 WMMA)
#pragma unroll
    for (int j = 0; j < 4; j++) {
      v8f s = vzero;
#pragma unroll
      for (int kc = 0; kc < 64; kc += 32) {
        v16bf a  = frag_a (&Qs[0][0], 72, wid * 16, kc, lane);
        v16bf bb = frag_bT(&Ks[0][0], 72, j * 16,  kc, lane);
        s = __builtin_amdgcn_wmma_f32_16x16x32_bf16(false, a, false, bb,
                                                    (short)0, s, false, false);
      }
#pragma unroll
      for (int r = 0; r < 8; r++)
        Ss[wid * 16 + half * 8 + r][j * 16 + ln16] = s[r] * scale;
    }
    __syncthreads();

    // online softmax row update (one thread per query row)
    if (tid < 128) {
      int row = tid;
      float mb = -3.0e38f;
      for (int c = 0; c < 64; c++) mb = fmaxf(mb, Ss[row][c]);
      float mold = rowmax[row];
      float mnew = fmaxf(mold, mb);
      float alpha = __expf(mold - mnew);     // 0 on first block
      float sum = 0.f;
      for (int c = 0; c < 64; c++) {
        float p = __expf(Ss[row][c] - mnew);
        Ps[row][c] = f2bf(p);
        sum += p;
      }
      rowsum[row]   = rowsum[row] * alpha + sum;
      rowmax[row]   = mnew;
      rowalpha[row] = alpha;
    }
    __syncthreads();

    // rescale running O, accumulate P@V (keys=64 -> 2 WMMA per 16-col tile)
    float al[8];
#pragma unroll
    for (int r = 0; r < 8; r++) al[r] = rowalpha[wid * 16 + half * 8 + r];
#pragma unroll
    for (int j = 0; j < 4; j++) {
#pragma unroll
      for (int r = 0; r < 8; r++) oacc[j][r] *= al[r];
#pragma unroll
      for (int kc = 0; kc < 64; kc += 32) {
        v16bf a  = frag_a  (&Ps[0][0], 72, wid * 16, kc, lane);
        v16bf bb = frag_bRM(&Vs[0][0], 72, kc, j * 16, lane);
        oacc[j] = __builtin_amdgcn_wmma_f32_16x16x32_bf16(false, a, false, bb,
                                                          (short)0, oacc[j],
                                                          false, false);
      }
    }
  }
  __syncthreads();

  float inv[8];
#pragma unroll
  for (int r = 0; r < 8; r++) inv[r] = 1.0f / rowsum[wid * 16 + half * 8 + r];
#pragma unroll
  for (int j = 0; j < 4; j++) {
#pragma unroll
    for (int r = 0; r < 8; r++) {
      int rowg = qbase + wid * 16 + half * 8 + r;
      int col  = j * 16 + ln16;
      O[base + (size_t)rowg * D_MODEL + col] = f2bf(oacc[j][r] * inv[r]);
    }
  }
}

// ---------------------------------------------------------------------------

extern "C" void kernel_launch(void* const* d_in, const int* in_sizes, int n_in,
                              void* d_out, int out_size, void* d_ws, size_t ws_size,
                              hipStream_t stream) {
  const float* x    = (const float*)d_in[0];
  const float* wq   = (const float*)d_in[1];
  const float* wk   = (const float*)d_in[2];
  const float* wv   = (const float*)d_in[3];
  const float* wo   = (const float*)d_in[4];
  const float* ssc  = (const float*)d_in[5];
  const float* w0   = (const float*)d_in[6];
  const float* b0   = (const float*)d_in[7];
  const float* w1   = (const float*)d_in[8];
  const float* b1   = (const float*)d_in[9];
  const float* ln1s = (const float*)d_in[10];
  const float* ln1b = (const float*)d_in[11];
  const float* ln2s = (const float*)d_in[12];
  const float* ln2b = (const float*)d_in[13];

  // ---- workspace carve-out (~128 MB total) ----
  char* ws = (char*)d_ws;
  size_t off = 0;
  auto carve = [&](size_t bytes) -> void* {
    void* p = ws + off;
    off += (bytes + 255) & ~(size_t)255;
    return p;
  };
  const size_t DD  = (size_t)D_MODEL * D_MODEL;
  const size_t DF  = (size_t)D_MODEL * D_FF;
  const size_t RD  = (size_t)ROWS * D_MODEL;
  const size_t RF  = (size_t)ROWS * D_FF;

  unsigned short* xb   = (unsigned short*)carve(RD * 2);  // bf16(x)
  unsigned short* xnb  = (unsigned short*)carve(RD * 2);  // bf16(LN1(x))
  unsigned short* wqb  = (unsigned short*)carve(DD * 2);
  unsigned short* wkb  = (unsigned short*)carve(DD * 2);
  unsigned short* wvb  = (unsigned short*)carve(DD * 2);
  unsigned short* wob  = (unsigned short*)carve(DD * 2);
  unsigned short* w0b  = (unsigned short*)carve(DF * 2);
  unsigned short* w1b  = (unsigned short*)carve(DF * 2);
  unsigned short* qb   = (unsigned short*)carve(RD * 2);
  unsigned short* kbuf = (unsigned short*)carve(RD * 2);
  unsigned short* vbuf = (unsigned short*)carve(RD * 2);
  unsigned short* ob   = (unsigned short*)carve(RD * 2);  // attn out, bf16
  float*          x1   = (float*)carve(RD * 4);           // x + attn@wo
  unsigned short* l2b  = (unsigned short*)carve(RD * 2);  // bf16(LN2(x1))
  unsigned short* hb   = (unsigned short*)carve(RF * 2);  // gelu(mlp mid)

  // ---- 1) weight casts ----
  {
    int n = (int)DD;
    int g = (n / 4 + 255) / 256;
    cast_f32_bf16_kernel<<<g, 256, 0, stream>>>(wq, wqb, n);
    cast_f32_bf16_kernel<<<g, 256, 0, stream>>>(wk, wkb, n);
    cast_f32_bf16_kernel<<<g, 256, 0, stream>>>(wv, wvb, n);
    cast_f32_bf16_kernel<<<g, 256, 0, stream>>>(wo, wob, n);
    int nf = (int)DF;
    int gf = (nf / 4 + 255) / 256;
    cast_f32_bf16_kernel<<<gf, 256, 0, stream>>>(w0, w0b, nf);
    cast_f32_bf16_kernel<<<gf, 256, 0, stream>>>(w1, w1b, nf);
  }

  // ---- 2) LN1 + cast x ----
  layernorm_cast_kernel<<<ROWS, 256, 0, stream>>>(x, ln1s, ln1b, xnb, xb);

  // ---- 3) Q/K/V projections (bf16 out) ----
  dim3 gD(D_MODEL / 128, ROWS / 128);   // (8, 32)
  gemm_bf16_kernel<<<gD, 256, 0, stream>>>(xnb, wqb, ROWS, D_MODEL, D_MODEL,
                                           nullptr, nullptr, nullptr, qb, 0);
  gemm_bf16_kernel<<<gD, 256, 0, stream>>>(xb, wkb, ROWS, D_MODEL, D_MODEL,
                                           nullptr, nullptr, nullptr, kbuf, 0);
  gemm_bf16_kernel<<<gD, 256, 0, stream>>>(xb, wvb, ROWS, D_MODEL, D_MODEL,
                                           nullptr, nullptr, nullptr, vbuf, 0);

  // ---- 4) attention ----
  flash_attn_kernel<<<BATCH * N_HEADS * (SEQ / 128), 256, 0, stream>>>(
      qb, kbuf, vbuf, ob, ssc);

  // ---- 5) output projection + residual: x1 = x + attn @ wo ----
  gemm_bf16_kernel<<<gD, 256, 0, stream>>>(ob, wob, ROWS, D_MODEL, D_MODEL,
                                           nullptr, x, x1, nullptr, 0);

  // ---- 6) LN2 ----
  layernorm_cast_kernel<<<ROWS, 256, 0, stream>>>(x1, ln2s, ln2b, l2b, nullptr);

  // ---- 7) MLP up + GELU (bf16 out) ----
  dim3 gF(D_FF / 128, ROWS / 128);      // (32, 32)
  gemm_bf16_kernel<<<gF, 256, 0, stream>>>(l2b, w0b, ROWS, D_FF, D_MODEL,
                                           b0, nullptr, nullptr, hb, 1);

  // ---- 8) MLP down + bias + residual -> d_out (fp32) ----
  gemm_bf16_kernel<<<gD, 256, 0, stream>>>(hb, w1b, ROWS, D_MODEL, D_FF,
                                           b1, x1, (float*)d_out, nullptr, 0);
}
